// Seq2seq_80513456931303
// MI455X (gfx1250) — matrix operand order repaired and checked by reading, compiled, not run on hardware
//
#include <hip/hip_runtime.h>
#include <hip/hip_bf16.h>

typedef __attribute__((ext_vector_type(16))) __bf16 v16bf;
typedef __attribute__((ext_vector_type(8)))  float  v8f;

#define U_DIM 1024
#define E_DIM 256
#define V_DIM 20200
#define B_DIM 32
#define S_DIM 64
#define T_DIM 16
#define N3U   3072          // 3*U
#define NT3U  192           // 3*U/16 column tiles
#define VT    1263          // ceil(V/16) column tiles (pad 20200 -> 20208)

// ---------- WMMA fragment helpers ----------

static __device__ __forceinline__ v16bf make_v16bf(uint4 lo, uint4 hi) {
  union { uint4 u[2]; v16bf v; } x;
  x.u[0] = lo; x.u[1] = hi;
  return x.v;
}

// A (16x32 bf16, row-major source [M x K]):
// lane<16: row m0+lane, K = k0+0..7 (v0..3) and k0+16..23 (v4..7)
// lane>=16: row m0+lane-16, K = k0+8..15 and k0+24..31
static __device__ __forceinline__ v16bf load_a_frag(const __bf16* A, int lda, int m0, int k0, int lane) {
  const __bf16* p = A + (size_t)(m0 + (lane & 15)) * lda + k0 + ((lane >> 4) << 3);
  uint4 lo = *(const uint4*)p;
  uint4 hi = *(const uint4*)(p + 16);
  return make_v16bf(lo, hi);
}

// B pre-packed: per (nt,kt) tile, 32 lanes x 16 bf16 contiguous (32B/lane)
static __device__ __forceinline__ v16bf load_b_frag(const __bf16* Bp, int nt, int kt, int Ktiles, int lane) {
  const __bf16* p = Bp + ((((size_t)nt * Ktiles + kt) * 32 + lane) << 4);
  uint4 lo = *(const uint4*)p;
  uint4 hi = *(const uint4*)(p + 8);
  return make_v16bf(lo, hi);
}

static __device__ __forceinline__ v8f wmma_bf16(v16bf a, v16bf b, v8f c) {
  return __builtin_amdgcn_wmma_f32_16x16x32_bf16(false, a, false, b, (short)0, c, false, false);
}

static __device__ __forceinline__ float sigmoidf_(float x) { return 1.0f / (1.0f + __expf(-x)); }

static __device__ __forceinline__ int imin_(int a, int b) { return a < b ? a : b; }

// ---------- weight pre-swizzle (fp32 row-major [K x Nsrc] -> packed bf16 B-fragments) ----------
__global__ void pack_b_kernel(const float* __restrict__ src, __bf16* __restrict__ dst,
                              int Nsrc, int Ntiles, int Ktiles) {
  size_t id = (size_t)blockIdx.x * blockDim.x + threadIdx.x;
  size_t total = (size_t)Ntiles * Ktiles * 512;   // 32 lanes * 16 elems per tile
  if (id >= total) return;
  int e    = (int)(id & 15);
  int lane = (int)((id >> 4) & 31);
  size_t tile = id >> 9;
  int kt = (int)(tile % Ktiles);
  int nt = (int)(tile / Ktiles);
  int k = kt * 32 + ((lane >> 4) << 3) + (e & 7) + ((e >> 3) << 4);   // lane<16: K=kt*32+{0..7,16..23} ; lane>=16: +8
  int n = nt * 16 + (lane & 15);
  float v = (n < Nsrc) ? src[(size_t)k * Nsrc + n] : 0.0f;
  dst[id] = (__bf16)v;
}

// ---------- encoder embedding gather: xA[t*32+b][e] = Ee[enc_in[b][t]][e] (bf16) ----------
__global__ void enc_embed_kernel(const int* __restrict__ enc_in, const float* __restrict__ Ee,
                                 __bf16* __restrict__ xA) {
  size_t id = (size_t)blockIdx.x * blockDim.x + threadIdx.x;   // 2048*256
  int e = (int)(id & 255);
  int row = (int)(id >> 8);
  int t = row >> 5;
  int b = row & 31;
  int tok = enc_in[b * S_DIM + t];
  xA[id] = (__bf16)Ee[(size_t)tok * E_DIM + e];
}

// ---------- generic bf16 WMMA GEMM: C[M,N] = A[M,K] @ B + bias ----------
// block = 256 threads = 8 waves (2 in M x 4 in N); wave tile = 16M x 64N
// Out-of-range tiles are CLAMPED for loads (WMMA always executes, EXEC all-1s);
// stores are bounds-checked in the epilogue only.
__global__ void __launch_bounds__(256)
gemm_bf16_kernel(const __bf16* __restrict__ A, int lda,
                 const __bf16* __restrict__ Bp,
                 const float* __restrict__ bias,
                 float* __restrict__ C, long long ldc,
                 int Ktiles, int Ntiles, int Nreal) {
  int lane = threadIdx.x & 31;
  int wave = threadIdx.x >> 5;
  int m0 = blockIdx.y * 32 + (wave >> 2) * 16;
  int ntbase = blockIdx.x * 16 + (wave & 3) * 4;
  int mx = Ntiles - 1;
  int nt0 = imin_(ntbase + 0, mx);
  int nt1 = imin_(ntbase + 1, mx);
  int nt2 = imin_(ntbase + 2, mx);
  int nt3 = imin_(ntbase + 3, mx);

  v8f acc0 = {}, acc1 = {}, acc2 = {}, acc3 = {};

  // software pipeline: prefetch k+1 fragments before issuing k's WMMAs
  v16bf a  = load_a_frag(A, lda, m0, 0, lane);
  v16bf b0 = load_b_frag(Bp, nt0, 0, Ktiles, lane);
  v16bf b1 = load_b_frag(Bp, nt1, 0, Ktiles, lane);
  v16bf b2 = load_b_frag(Bp, nt2, 0, Ktiles, lane);
  v16bf b3 = load_b_frag(Bp, nt3, 0, Ktiles, lane);
  for (int kt = 0; kt + 1 < Ktiles; ++kt) {
    v16bf an  = load_a_frag(A, lda, m0, (kt + 1) * 32, lane);
    v16bf bn0 = load_b_frag(Bp, nt0, kt + 1, Ktiles, lane);
    v16bf bn1 = load_b_frag(Bp, nt1, kt + 1, Ktiles, lane);
    v16bf bn2 = load_b_frag(Bp, nt2, kt + 1, Ktiles, lane);
    v16bf bn3 = load_b_frag(Bp, nt3, kt + 1, Ktiles, lane);
    acc0 = wmma_bf16(a, b0, acc0);
    acc1 = wmma_bf16(a, b1, acc1);
    acc2 = wmma_bf16(a, b2, acc2);
    acc3 = wmma_bf16(a, b3, acc3);
    a = an; b0 = bn0; b1 = bn1; b2 = bn2; b3 = bn3;
  }
  acc0 = wmma_bf16(a, b0, acc0);
  acc1 = wmma_bf16(a, b1, acc1);
  acc2 = wmma_bf16(a, b2, acc2);
  acc3 = wmma_bf16(a, b3, acc3);

  int rbase = m0 + ((lane >> 4) << 3);
  int ci = lane & 15;
  v8f accs[4] = {acc0, acc1, acc2, acc3};
  for (int t = 0; t < 4; ++t) {
    int nt = ntbase + t;
    if (nt >= Ntiles) continue;
    int n = nt * 16 + ci;
    if (n >= Nreal) continue;
    float bb = bias ? bias[n] : 0.0f;
    v8f acv = accs[t];
    for (int r = 0; r < 8; ++r)
      C[(size_t)(rbase + r) * ldc + n] = acv[r] + bb;
  }
}

// ---------- fused encoder GRU step: gh = h@Wh, then gates ----------
// grid 16 x (block 128 = 4 waves); wave handles one 16-col j-tile across 3 gates, both M halves
__global__ void __launch_bounds__(128)
enc_step_kernel(const __bf16* __restrict__ hA,     // [32,1024] bf16
                const float* __restrict__ h_in,    // [32,1024] f32
                const __bf16* __restrict__ WhP,    // packed, Ktiles=32, Ntiles=192
                const float* __restrict__ gx_t,    // [32,3072] = x@Wx + b_in
                const float* __restrict__ b_rec,   // [3072]
                float* __restrict__ h_out,
                __bf16* __restrict__ hA_out,
                float* __restrict__ enc_f32_t,     // rows t*32.., [32,1024]
                __bf16* __restrict__ enc_bf16_t) {
  const int Ktiles = 32;
  int lane = threadIdx.x & 31;
  int wave = threadIdx.x >> 5;
  int jt = blockIdx.x * 4 + wave;          // 0..63

  v8f az0 = {}, az1 = {}, ar0 = {}, ar1 = {}, ah0 = {}, ah1 = {};

  v16bf a0 = load_a_frag(hA, U_DIM, 0, 0, lane);
  v16bf a1 = load_a_frag(hA, U_DIM, 16, 0, lane);
  v16bf bz = load_b_frag(WhP, jt, 0, Ktiles, lane);
  v16bf br = load_b_frag(WhP, 64 + jt, 0, Ktiles, lane);
  v16bf bh = load_b_frag(WhP, 128 + jt, 0, Ktiles, lane);
  for (int kt = 0; kt + 1 < Ktiles; ++kt) {
    v16bf na0 = load_a_frag(hA, U_DIM, 0, (kt + 1) * 32, lane);
    v16bf na1 = load_a_frag(hA, U_DIM, 16, (kt + 1) * 32, lane);
    v16bf nbz = load_b_frag(WhP, jt, kt + 1, Ktiles, lane);
    v16bf nbr = load_b_frag(WhP, 64 + jt, kt + 1, Ktiles, lane);
    v16bf nbh = load_b_frag(WhP, 128 + jt, kt + 1, Ktiles, lane);
    az0 = wmma_bf16(a0, bz, az0);  az1 = wmma_bf16(a1, bz, az1);
    ar0 = wmma_bf16(a0, br, ar0);  ar1 = wmma_bf16(a1, br, ar1);
    ah0 = wmma_bf16(a0, bh, ah0);  ah1 = wmma_bf16(a1, bh, ah1);
    a0 = na0; a1 = na1; bz = nbz; br = nbr; bh = nbh;
  }
  az0 = wmma_bf16(a0, bz, az0);  az1 = wmma_bf16(a1, bz, az1);
  ar0 = wmma_bf16(a0, br, ar0);  ar1 = wmma_bf16(a1, br, ar1);
  ah0 = wmma_bf16(a0, bh, ah0);  ah1 = wmma_bf16(a1, bh, ah1);

  int j = jt * 16 + (lane & 15);
  float brz = b_rec[j], brr = b_rec[j + U_DIM], brh = b_rec[j + 2 * U_DIM];
  for (int half = 0; half < 2; ++half) {
    v8f az = half ? az1 : az0;
    v8f ar = half ? ar1 : ar0;
    v8f ah = half ? ah1 : ah0;
    int rbase = half * 16 + ((lane >> 4) << 3);
    for (int r = 0; r < 8; ++r) {
      int row = rbase + r;
      const float* gx = gx_t + (size_t)row * N3U;
      float z  = sigmoidf_(gx[j]             + az[r] + brz);
      float rr = sigmoidf_(gx[j + U_DIM]     + ar[r] + brr);
      float c  = tanhf   (gx[j + 2 * U_DIM] + rr * (ah[r] + brh));
      float hp = h_in[(size_t)row * U_DIM + j];
      float hn = z * hp + (1.0f - z) * c;
      size_t o = (size_t)row * U_DIM + j;
      h_out[o] = hn;
      __bf16 hb = (__bf16)hn;
      hA_out[o] = hb;
      enc_f32_t[o] = hn;
      enc_bf16_t[o] = hb;
    }
  }
}

// ---------- attention: scores, softmax, context, build x = [ctx, Ed[tok]] (bf16) ----------
__global__ void attn_kernel(const float* __restrict__ pre_enc,  // [64*32,1024], row = s*32+b
                            const float* __restrict__ q,        // [32,1024]
                            const float* __restrict__ enc_f32,  // [64*32,1024]
                            const float* __restrict__ Va,       // [1024]
                            const float* __restrict__ bv,       // [1]
                            const float* __restrict__ Ed,       // [V,256]
                            const int* __restrict__ teacher,    // [32,16]
                            int t,
                            __bf16* __restrict__ xA) {          // [32,1280]
  int b = blockIdx.x;
  int tid = threadIdx.x;   // 256
  __shared__ float red[256];
  __shared__ float sc[S_DIM];

  const float* qb = q + (size_t)b * U_DIM;
  for (int s = 0; s < S_DIM; ++s) {
    const float* pe = pre_enc + (size_t)(s * 32 + b) * U_DIM;
    float p = 0.0f;
    for (int u = tid; u < U_DIM; u += 256)
      p += tanhf(pe[u] + qb[u]) * Va[u];
    red[tid] = p;
    __syncthreads();
    for (int o = 128; o > 0; o >>= 1) {
      if (tid < o) red[tid] += red[tid + o];
      __syncthreads();
    }
    if (tid == 0) sc[s] = red[0] + bv[0];
    __syncthreads();
  }

  if (tid == 0) {
    float m = sc[0];
    for (int s = 1; s < S_DIM; ++s) m = fmaxf(m, sc[s]);
    float sum = 0.0f;
    for (int s = 0; s < S_DIM; ++s) { sc[s] = __expf(sc[s] - m); sum += sc[s]; }
    float inv = 1.0f / sum;
    for (int s = 0; s < S_DIM; ++s) sc[s] *= inv;
  }
  __syncthreads();

  for (int u = tid; u < U_DIM; u += 256) {
    float a = 0.0f;
    for (int s = 0; s < S_DIM; ++s)
      a += sc[s] * enc_f32[(size_t)(s * 32 + b) * U_DIM + u];
    xA[(size_t)b * (U_DIM + E_DIM) + u] = (__bf16)a;
  }
  int tok = teacher[b * T_DIM + t];
  for (int e = tid; e < E_DIM; e += 256)
    xA[(size_t)b * (U_DIM + E_DIM) + U_DIM + e] = (__bf16)Ed[(size_t)tok * E_DIM + e];
}

// ---------- fused decoder GRU step: gx = x@Wx, gh = b_rec (h0 == 0), h_new = (1-z)*c ----------
__global__ void __launch_bounds__(128)
dec_gate_kernel(const __bf16* __restrict__ xA,   // [32,1280] bf16
                const __bf16* __restrict__ WxP,  // packed, Ktiles=40, Ntiles=192
                const float* __restrict__ b_in,  // [3072]
                const float* __restrict__ b_rec, // [3072]
                __bf16* __restrict__ h_out) {    // [32,1024] bf16
  const int Ktiles = 40;
  const int lda = U_DIM + E_DIM;
  int lane = threadIdx.x & 31;
  int wave = threadIdx.x >> 5;
  int jt = blockIdx.x * 4 + wave;

  v8f az0 = {}, az1 = {}, ar0 = {}, ar1 = {}, ah0 = {}, ah1 = {};

  v16bf a0 = load_a_frag(xA, lda, 0, 0, lane);
  v16bf a1 = load_a_frag(xA, lda, 16, 0, lane);
  v16bf bz = load_b_frag(WxP, jt, 0, Ktiles, lane);
  v16bf br = load_b_frag(WxP, 64 + jt, 0, Ktiles, lane);
  v16bf bh = load_b_frag(WxP, 128 + jt, 0, Ktiles, lane);
  for (int kt = 0; kt + 1 < Ktiles; ++kt) {
    v16bf na0 = load_a_frag(xA, lda, 0, (kt + 1) * 32, lane);
    v16bf na1 = load_a_frag(xA, lda, 16, (kt + 1) * 32, lane);
    v16bf nbz = load_b_frag(WxP, jt, kt + 1, Ktiles, lane);
    v16bf nbr = load_b_frag(WxP, 64 + jt, kt + 1, Ktiles, lane);
    v16bf nbh = load_b_frag(WxP, 128 + jt, kt + 1, Ktiles, lane);
    az0 = wmma_bf16(a0, bz, az0);  az1 = wmma_bf16(a1, bz, az1);
    ar0 = wmma_bf16(a0, br, ar0);  ar1 = wmma_bf16(a1, br, ar1);
    ah0 = wmma_bf16(a0, bh, ah0);  ah1 = wmma_bf16(a1, bh, ah1);
    a0 = na0; a1 = na1; bz = nbz; br = nbr; bh = nbh;
  }
  az0 = wmma_bf16(a0, bz, az0);  az1 = wmma_bf16(a1, bz, az1);
  ar0 = wmma_bf16(a0, br, ar0);  ar1 = wmma_bf16(a1, br, ar1);
  ah0 = wmma_bf16(a0, bh, ah0);  ah1 = wmma_bf16(a1, bh, ah1);

  int j = jt * 16 + (lane & 15);
  float biz = b_in[j], bir = b_in[j + U_DIM], bih = b_in[j + 2 * U_DIM];
  float brz = b_rec[j], brr = b_rec[j + U_DIM], brh = b_rec[j + 2 * U_DIM];
  for (int half = 0; half < 2; ++half) {
    v8f az = half ? az1 : az0;
    v8f ar = half ? ar1 : ar0;
    v8f ah = half ? ah1 : ah0;
    int rbase = half * 16 + ((lane >> 4) << 3);
    for (int r = 0; r < 8; ++r) {
      int row = rbase + r;
      float z  = sigmoidf_(az[r] + biz + brz);
      float rr = sigmoidf_(ar[r] + bir + brr);
      float c  = tanhf(ah[r] + bih + rr * brh);
      float hn = (1.0f - z) * c;
      h_out[(size_t)row * U_DIM + j] = (__bf16)hn;
    }
  }
}

// ---------- argmax over V per (b,t); predictions as int64 after logits ----------
__global__ void argmax_kernel(const float* __restrict__ logits, long long* __restrict__ pred) {
  int bt = blockIdx.x;
  int tid = threadIdx.x;  // 256
  const float* p = logits + (size_t)bt * V_DIM;
  float best = -__builtin_huge_valf();
  int bi = V_DIM;
  for (int i = tid; i < V_DIM; i += 256) {
    float v = p[i];
    if (v > best) { best = v; bi = i; }
  }
  __shared__ float bvs[256];
  __shared__ int bis[256];
  bvs[tid] = best; bis[tid] = bi;
  __syncthreads();
  for (int o = 128; o > 0; o >>= 1) {
    if (tid < o) {
      if (bvs[tid + o] > bvs[tid] ||
          (bvs[tid + o] == bvs[tid] && bis[tid + o] < bis[tid])) {
        bvs[tid] = bvs[tid + o]; bis[tid] = bis[tid + o];
      }
    }
    __syncthreads();
  }
  if (tid == 0) pred[bt] = (long long)bis[0];
}

// ---------- host ----------
extern "C" void kernel_launch(void* const* d_in, const int* in_sizes, int n_in,
                              void* d_out, int out_size, void* d_ws, size_t ws_size,
                              hipStream_t stream) {
  const int*   enc_in    = (const int*)d_in[0];
  const int*   teacher   = (const int*)d_in[1];
  const float* Ee        = (const float*)d_in[4];
  const float* enc_Wx    = (const float*)d_in[5];
  const float* enc_Wh    = (const float*)d_in[6];
  const float* enc_b_in  = (const float*)d_in[7];
  const float* enc_b_rec = (const float*)d_in[8];
  const float* Ed        = (const float*)d_in[9];
  const float* dec_Wx    = (const float*)d_in[10];
  const float* dec_b_in  = (const float*)d_in[12];
  const float* dec_b_rec = (const float*)d_in[13];
  const float* W1 = (const float*)d_in[14];
  const float* b1 = (const float*)d_in[15];
  const float* W2 = (const float*)d_in[16];
  const float* b2 = (const float*)d_in[17];
  const float* Va = (const float*)d_in[18];
  const float* bvp = (const float*)d_in[19];
  const float* Wf = (const float*)d_in[20];
  const float* bf = (const float*)d_in[21];

  char* ws = (char*)d_ws;
  size_t off = 0;
  auto alloc = [&](size_t bytes) -> char* {
    char* p = ws + off;
    off = (off + bytes + 255) & ~(size_t)255;
    return p;
  };

  // packed weights (bf16): tile = Ktiles*Ntiles, 1024 B per tile
  __bf16* encWxP = (__bf16*)alloc((size_t)8  * NT3U * 1024);
  __bf16* encWhP = (__bf16*)alloc((size_t)32 * NT3U * 1024);
  __bf16* decWxP = (__bf16*)alloc((size_t)40 * NT3U * 1024);
  __bf16* W1P    = (__bf16*)alloc((size_t)32 * 64 * 1024);
  __bf16* W2P    = (__bf16*)alloc((size_t)32 * 64 * 1024);
  __bf16* WfP    = (__bf16*)alloc((size_t)32 * VT * 1024);
  // activations
  __bf16* xEnc   = (__bf16*)alloc((size_t)2048 * E_DIM * 2);       // [t*32+b, E]
  float*  gxAll  = (float*) alloc((size_t)2048 * N3U * 4);         // [t*32+b, 3U]
  float*  hF[2]  = {(float*)alloc((size_t)B_DIM * U_DIM * 4),
                    (float*)alloc((size_t)B_DIM * U_DIM * 4)};
  __bf16* hB[2]  = {(__bf16*)alloc((size_t)B_DIM * U_DIM * 2),
                    (__bf16*)alloc((size_t)B_DIM * U_DIM * 2)};
  float*  encF   = (float*) alloc((size_t)2048 * U_DIM * 4);       // [s*32+b, U]
  __bf16* encB   = (__bf16*)alloc((size_t)2048 * U_DIM * 2);
  float*  preEnc = (float*) alloc((size_t)2048 * U_DIM * 4);
  float*  qBuf   = (float*) alloc((size_t)B_DIM * U_DIM * 4);
  __bf16* xDec   = (__bf16*)alloc((size_t)B_DIM * (U_DIM + E_DIM) * 2);

  auto packLaunch = [&](const float* src, __bf16* dst, int Nsrc, int Ntiles, int Ktiles) {
    size_t total = (size_t)Ntiles * Ktiles * 512;
    pack_b_kernel<<<dim3((unsigned)((total + 255) / 256)), dim3(256), 0, stream>>>(src, dst, Nsrc, Ntiles, Ktiles);
  };
  packLaunch(enc_Wx, encWxP, N3U, NT3U, 8);
  packLaunch(enc_Wh, encWhP, N3U, NT3U, 32);
  packLaunch(dec_Wx, decWxP, N3U, NT3U, 40);
  packLaunch(W1, W1P, U_DIM, 64, 32);
  packLaunch(W2, W2P, U_DIM, 64, 32);
  packLaunch(Wf, WfP, V_DIM, VT, 32);

  // encoder embeddings + batched gx_all = x @ enc_Wx + b_in
  enc_embed_kernel<<<dim3(2048), dim3(256), 0, stream>>>(enc_in, Ee, xEnc);
  gemm_bf16_kernel<<<dim3((NT3U + 15) / 16, 2048 / 32), dim3(256), 0, stream>>>(
      xEnc, E_DIM, encWxP, enc_b_in, gxAll, (long long)N3U, 8, NT3U, N3U);

  // h0 = 0
  hipMemsetAsync(hF[0], 0, (size_t)B_DIM * U_DIM * 4, stream);
  hipMemsetAsync(hB[0], 0, (size_t)B_DIM * U_DIM * 2, stream);

  // encoder recurrence (64 fused GEMM+gate steps, ping-pong h)
  for (int t = 0; t < S_DIM; ++t) {
    int in = t & 1, out = (t + 1) & 1;
    enc_step_kernel<<<dim3(16), dim3(128), 0, stream>>>(
        hB[in], hF[in], encWhP,
        gxAll + (size_t)t * 32 * N3U, enc_b_rec,
        hF[out], hB[out],
        encF + (size_t)t * 32 * U_DIM,
        encB + (size_t)t * 32 * U_DIM);
  }
  __bf16* decH = hB[0];   // after 64 steps the final hidden lives in buffer 0

  // pre_enc = encoded @ W1 + b1
  gemm_bf16_kernel<<<dim3(4, 2048 / 32), dim3(256), 0, stream>>>(
      encB, U_DIM, W1P, b1, preEnc, (long long)U_DIM, 32, 64, U_DIM);

  float* logits = (float*)d_out;
  const long long ldLogits = (long long)T_DIM * V_DIM;

  // decoder loop
  for (int t = 0; t < T_DIM; ++t) {
    // q = dec_h @ W2 + b2
    gemm_bf16_kernel<<<dim3(4, 1), dim3(256), 0, stream>>>(
        decH, U_DIM, W2P, b2, qBuf, (long long)U_DIM, 32, 64, U_DIM);
    // attention + context + x build
    attn_kernel<<<dim3(B_DIM), dim3(256), 0, stream>>>(
        preEnc, qBuf, encF, Va, bvp, Ed, teacher, t, xDec);
    // gx = x @ dec_Wx, gates (h0 = 0) -> new dec_h (bf16)
    dec_gate_kernel<<<dim3(16), dim3(128), 0, stream>>>(
        xDec, decWxP, dec_b_in, dec_b_rec, decH);
    // logits_t = h_new @ Wf + bf  -> d_out[b][t][:]
    gemm_bf16_kernel<<<dim3((VT + 15) / 16, 1), dim3(256), 0, stream>>>(
        decH, U_DIM, WfP, bf, logits + (size_t)t * V_DIM, ldLogits, 32, VT, V_DIM);
  }

  // predictions (int64) appended after logits
  long long* pred = (long long*)(logits + (size_t)B_DIM * T_DIM * V_DIM);
  argmax_kernel<<<dim3(B_DIM * T_DIM), dim3(256), 0, stream>>>(logits, pred);
}
